// LoRAAttention_91061896609820
// MI455X (gfx1250) — matrix operand order repaired and checked by reading, not compiled, not run on hardware
//
#include <hip/hip_runtime.h>
#include <hip/hip_bf16.h>
#include <stdint.h>

// ---------------------------------------------------------------------------
// Problem constants (from reference)
// ---------------------------------------------------------------------------
constexpr int N_B  = 4;      // batch
constexpr int L_S  = 1024;   // sequence length
constexpr int C_D  = 1024;   // model dim
constexpr int H_N  = 16;     // heads
constexpr int HD   = 64;     // head dim
constexpr int R_LR = 8;      // LoRA rank
constexpr float LORA_SCALE = 2.0f;               // 16/8
constexpr float LOGIT_MAX  = 4.605170185988091f; // ln(100)

typedef __attribute__((ext_vector_type(16))) __bf16 v16bf;
typedef __attribute__((ext_vector_type(8)))  __bf16 v8bf;
typedef __attribute__((ext_vector_type(8)))  float  v8f;

// ---------------------------------------------------------------------------
// WMMA helper: gfx1250 wave32 16x16x32 bf16 -> f32
// ---------------------------------------------------------------------------
__device__ __forceinline__ v8f wmma_bf16(v16bf a, v16bf b, v8f c) {
  return __builtin_amdgcn_wmma_f32_16x16x32_bf16(
      /*neg_a=*/false, a, /*neg_b=*/false, b,
      /*c_mod=*/(short)0, c, /*reuse_a=*/false, /*reuse_b=*/false);
}

// ---------------------------------------------------------------------------
// CDNA5 async global->LDS copy (ASYNCcnt-tracked), ISA §15.18.3 op 98.
//   vdst VGPR = per-lane LDS byte address, vaddr = 64-bit global address.
// ---------------------------------------------------------------------------
__device__ __forceinline__ void async_b128(unsigned lds_off, const void* gptr) {
  uint64_t ga = (uint64_t)(uintptr_t)gptr;
  asm volatile("global_load_async_to_lds_b128 %0, %1, off"
               :: "v"(lds_off), "v"(ga) : "memory");
}

template <int Nw>
__device__ __forceinline__ void wait_asynccnt() {
  asm volatile("s_wait_asynccnt %0" :: "n"(Nw) : "memory");
}

// ---------------------------------------------------------------------------
// Fragment loaders (work for both global and LDS source pointers).
// A-matrix fragment (16 rows x 32 K), row-major source, row stride lda.
// ---------------------------------------------------------------------------
__device__ __forceinline__ v16bf load_a_frag(const __bf16* A, int lda,
                                             int row0, int k0) {
  int lane = threadIdx.x & 31;
  int half = lane >> 4;
  int r    = lane & 15;
  const __bf16* p = A + (size_t)(row0 + r) * lda + k0 + half * 8;
  v8bf lo = *(const v8bf*)(p);
  v8bf hi = *(const v8bf*)(p + 16);
  v16bf a;
#pragma unroll
  for (int i = 0; i < 8; ++i) { a[i] = lo[i]; a[i + 8] = hi[i]; }
  return a;
}

// B-matrix fragment (32 K x 16 cols) from "transposed" storage Bt[n][k],
// row stride ldb (k contiguous per output column n).
__device__ __forceinline__ v16bf load_b_frag(const __bf16* Bt, int ldb,
                                             int n0, int k0) {
  int lane = threadIdx.x & 31;
  int half = lane >> 4;
  int n    = n0 + (lane & 15);
  const __bf16* p = Bt + (size_t)n * ldb + k0 + half * 8;
  v8bf lo = *(const v8bf*)(p);
  v8bf hi = *(const v8bf*)(p + 16);
  v16bf b;
#pragma unroll
  for (int i = 0; i < 8; ++i) { b[i] = lo[i]; b[i + 8] = hi[i]; }
  return b;
}

// ---------------------------------------------------------------------------
// 1) f32 -> bf16 convert (grid-stride)
// ---------------------------------------------------------------------------
__global__ void k_convert_bf16(const float* __restrict__ src,
                               __bf16* __restrict__ dst, int n) {
  for (int i = blockIdx.x * blockDim.x + threadIdx.x; i < n;
       i += gridDim.x * blockDim.x)
    dst[i] = (__bf16)src[i];
}

// ---------------------------------------------------------------------------
// 2) Fold LoRA into qkv weight:  W'[o][c] = W[o][c] + 2 * (B A)[o%C][c]
// ---------------------------------------------------------------------------
__global__ void k_merge_lora(const float* __restrict__ W,
                             const float* __restrict__ Aq, const float* __restrict__ Bq,
                             const float* __restrict__ Ak, const float* __restrict__ Bk,
                             const float* __restrict__ Av, const float* __restrict__ Bv,
                             __bf16* __restrict__ Wm) {
  int idx = blockIdx.x * blockDim.x + threadIdx.x;
  if (idx >= 3 * C_D * C_D) return;
  int o = idx / C_D;
  int c = idx - o * C_D;
  int sel = o / C_D;          // 0=q, 1=k, 2=v
  int oc  = o - sel * C_D;
  const float* A = (sel == 0) ? Aq : (sel == 1) ? Ak : Av; // [R, C]
  const float* B = (sel == 0) ? Bq : (sel == 1) ? Bk : Bv; // [C, R]
  float s = 0.0f;
#pragma unroll
  for (int r = 0; r < R_LR; ++r)
    s += B[oc * R_LR + r] * A[r * C_D + c];
  Wm[idx] = (__bf16)(W[idx] + LORA_SCALE * s);
}

// ---------------------------------------------------------------------------
// 3) WMMA GEMM with double-buffered async global->LDS staging.
//    D[M,Nn] = A[M,K] * Bt[Nn,K]^T + bias
//    Block: 128 threads (4 waves), tile M=32 x N=256, K-step 32.
// ---------------------------------------------------------------------------
constexpr int BM = 32;
constexpr int BN = 256;
constexpr int BK = 32;

__global__ void __launch_bounds__(128)
k_gemm_bf16(const __bf16* __restrict__ A, const __bf16* __restrict__ Bt,
            const float* __restrict__ bias, float* __restrict__ D,
            int M, int Nn, int K) {
  __shared__ __bf16 As[2][BM * BK];   //  2 KB x2
  __shared__ __bf16 Bs[2][BN * BK];   // 16 KB x2  -> 36 KB total

  int tid  = threadIdx.x;
  int wave = tid >> 5;
  int lane = tid & 31;
  int mblk = blockIdx.y * BM;
  int nblk = blockIdx.x * BN;

  int arow = tid >> 2, aseg = tid & 3;

  auto issue_stage = [&](int k0, int buf) {
    async_b128((unsigned)(uintptr_t)&As[buf][arow * BK + aseg * 8],
               A + (size_t)(mblk + arow) * K + k0 + aseg * 8);
#pragma unroll
    for (int i = 0; i < 8; ++i) {
      int id  = tid + i * 128;
      int row = id >> 2, seg = id & 3;
      async_b128((unsigned)(uintptr_t)&Bs[buf][row * BK + seg * 8],
                 Bt + (size_t)(nblk + row) * K + k0 + seg * 8);
    }
  };

  v8f acc[2][4] = {};
  const int nk = K / BK;

  issue_stage(0, 0);
  for (int i = 0; i < nk; ++i) {
    int buf = i & 1;
    if (i + 1 < nk) {
      issue_stage((i + 1) * BK, buf ^ 1);
      wait_asynccnt<9>();   // 9 issues/stage/wave, in-order -> stage i landed
    } else {
      wait_asynccnt<0>();
    }
    __syncthreads();

#pragma unroll
    for (int mi = 0; mi < 2; ++mi) {
      v16bf af = load_a_frag(As[buf], BK, mi * 16, 0);
#pragma unroll
      for (int t = 0; t < 4; ++t) {
        v16bf bfb = load_b_frag(Bs[buf], BK, wave * 64 + 16 * t, 0);
        acc[mi][t] = wmma_bf16(af, bfb, acc[mi][t]);
      }
    }
    __syncthreads();
  }

  int half  = lane >> 4;
  int cbase = lane & 15;
#pragma unroll
  for (int mi = 0; mi < 2; ++mi) {
    int rbase = mblk + mi * 16 + half * 8;
#pragma unroll
    for (int t = 0; t < 4; ++t) {
      int col = nblk + wave * 64 + 16 * t + cbase;
      float bv = bias ? bias[col] : 0.0f;
#pragma unroll
      for (int j = 0; j < 8; ++j)
        D[(size_t)(rbase + j) * Nn + col] = acc[mi][t][j] + bv;
    }
  }
}

// ---------------------------------------------------------------------------
// 4) Split qkv, l2-normalize q/k per (n,h,l), transpose v.
//    qkv f32 [N,L,3C] -> Qn,Kn bf16 [N,H,L,HD],  Vt bf16 [N,H,HD,L]
// ---------------------------------------------------------------------------
__global__ void k_norm_split(const float* __restrict__ qkv,
                             __bf16* __restrict__ Qn,
                             __bf16* __restrict__ Kn,
                             __bf16* __restrict__ Vt) {
  int id = blockIdx.x * blockDim.x + threadIdx.x; // over N*L*H
  if (id >= N_B * L_S * H_N) return;
  int h = id % H_N;
  int l = (id / H_N) % L_S;
  int n = id / (H_N * L_S);

  const float* base = qkv + ((size_t)(n * L_S + l)) * (3 * C_D) + h * HD;
  const float* q = base;
  const float* k = base + C_D;
  const float* v = base + 2 * C_D;

  float sq = 0.0f, sk = 0.0f;
#pragma unroll 8
  for (int d = 0; d < HD; ++d) { sq += q[d] * q[d]; sk += k[d] * k[d]; }
  float iq = 1.0f / fmaxf(sqrtf(sq), 1e-12f);
  float ik = 1.0f / fmaxf(sqrtf(sk), 1e-12f);

  size_t ohead = ((size_t)(n * H_N + h) * L_S + l) * HD;
  size_t vbase = ((size_t)(n * H_N + h) * HD) * L_S + l;
#pragma unroll 8
  for (int d = 0; d < HD; ++d) {
    Qn[ohead + d] = (__bf16)(q[d] * iq);
    Kn[ohead + d] = (__bf16)(k[d] * ik);
    Vt[vbase + (size_t)d * L_S] = (__bf16)v[d];
  }
}

// ---------------------------------------------------------------------------
// 5) Attention: one wave per (n,h, 16 query rows).
//    K and V tiles double-buffered into LDS via async copies (same-wave
//    ordering: s_wait_asynccnt -> ds reads; register deps force dscnt waits
//    before WMMA, ordering consumption before the next stage's overwrite).
// ---------------------------------------------------------------------------
__global__ void __launch_bounds__(32)
k_attention(const __bf16* __restrict__ Qn,
            const __bf16* __restrict__ Kn,
            const __bf16* __restrict__ Vt,
            const float* __restrict__ logit_scale,
            __bf16* __restrict__ Ao) {
  __shared__ float  S[16 * L_S];      // 64 KB score strip
  __shared__ __bf16 P[16 * L_S];      // 32 KB softmaxed probs
  __shared__ __bf16 Ks[2][16 * HD];   // 2 x 2 KB  K tiles (16 rows x 64 d)
  __shared__ __bf16 Vs[2][HD * BK];   // 2 x 4 KB  V tiles (64 d-rows x 32 k)

  int nh   = blockIdx.x;              // n*H + h
  int h    = nh & (H_N - 1);
  int n    = nh >> 4;
  int l0   = blockIdx.y * 16;
  int lane  = threadIdx.x & 31;
  int half  = lane >> 4;
  int cbase = lane & 15;

  const __bf16* Qh = Qn + (size_t)nh * L_S * HD;
  const __bf16* Kh = Kn + (size_t)nh * L_S * HD;
  const __bf16* Vh = Vt + (size_t)nh * HD * L_S;

  float scale = __expf(fminf(logit_scale[h], LOGIT_MAX));

  // K tile: 2 KB = 128 x 16B; 4 async issues per lane
  auto issue_k = [&](int m0, int buf) {
#pragma unroll
    for (int i = 0; i < 4; ++i) {
      int id = lane + i * 32;
      int row = id >> 3, seg = id & 7;
      async_b128((unsigned)(uintptr_t)&Ks[buf][row * HD + seg * 8],
                 Kh + (size_t)(m0 + row) * HD + seg * 8);
    }
  };
  // V tile: 4 KB = 256 x 16B; 8 async issues per lane
  auto issue_v = [&](int k0, int buf) {
#pragma unroll
    for (int i = 0; i < 8; ++i) {
      int id = lane + i * 32;
      int row = id >> 2, seg = id & 3;
      async_b128((unsigned)(uintptr_t)&Vs[buf][row * BK + seg * 8],
                 Vh + (size_t)row * L_S + k0 + seg * 8);
    }
  };

  // Q tile 16x64 as two A-fragments, held in VGPRs for the whole loop
  v16bf qa0 = load_a_frag(Qh, HD, l0, 0);
  v16bf qa1 = load_a_frag(Qh, HD, l0, 32);

  // --- S = scale * Qn Kn^T  (async-pipelined K tiles) ---------------------
  issue_k(0, 0);
  for (int m0 = 0; m0 < L_S; m0 += 16) {
    int buf = (m0 >> 4) & 1;
    if (m0 + 16 < L_S) { issue_k(m0 + 16, buf ^ 1); wait_asynccnt<4>(); }
    else               { wait_asynccnt<0>(); }
    v16bf kb0 = load_b_frag(Ks[buf], HD, 0, 0);
    v16bf kb1 = load_b_frag(Ks[buf], HD, 0, 32);
    v8f s = {};
    s = wmma_bf16(qa0, kb0, s);
    s = wmma_bf16(qa1, kb1, s);
    int col = m0 + cbase;
#pragma unroll
    for (int j = 0; j < 8; ++j)
      S[(half * 8 + j) * L_S + col] = s[j] * scale;
  }
  __syncthreads();

  // --- row softmax: lane pair (r, r+16) splits row r ----------------------
  {
    int r  = lane & 15;
    int i0 = half * (L_S / 2);
    float* row = S + r * L_S;
    float mx = -3.0e38f;
    for (int i = 0; i < L_S / 2; ++i) mx = fmaxf(mx, row[i0 + i]);
    mx = fmaxf(mx, __shfl_xor(mx, 16));
    float sum = 0.0f;
    for (int i = 0; i < L_S / 2; ++i) sum += __expf(row[i0 + i] - mx);
    sum += __shfl_xor(sum, 16);
    float inv = 1.0f / sum;
    __bf16* prow = P + r * L_S;
    for (int i = 0; i < L_S / 2; ++i)
      prow[i0 + i] = (__bf16)(__expf(row[i0 + i] - mx) * inv);
  }
  __syncthreads();

  // --- O = P @ V  (async-pipelined V tiles) -------------------------------
  v8f acc[4] = {};
  issue_v(0, 0);
  for (int k0 = 0; k0 < L_S; k0 += 32) {
    int buf = (k0 >> 5) & 1;
    if (k0 + 32 < L_S) { issue_v(k0 + 32, buf ^ 1); wait_asynccnt<8>(); }
    else               { wait_asynccnt<0>(); }
    v16bf pa = load_a_frag(P, L_S, 0, k0);
#pragma unroll
    for (int t = 0; t < 4; ++t) {
      v16bf vb = load_b_frag(Vs[buf], BK, 16 * t, 0);
      acc[t] = wmma_bf16(pa, vb, acc[t]);
    }
  }

  // store to Ao[n][l][h*64 + d]  (heads^-1 transpose fused)
  int rbase = l0 + half * 8;
#pragma unroll
  for (int t = 0; t < 4; ++t) {
    int d = 16 * t + cbase;
#pragma unroll
    for (int j = 0; j < 8; ++j)
      Ao[((size_t)n * L_S + rbase + j) * C_D + h * HD + d] = (__bf16)acc[t][j];
  }
}

// ---------------------------------------------------------------------------
// Host-side launch
// ---------------------------------------------------------------------------
extern "C" void kernel_launch(void* const* d_in, const int* in_sizes, int n_in,
                              void* d_out, int out_size, void* d_ws, size_t ws_size,
                              hipStream_t stream) {
  const float* x      = (const float*)d_in[0];   // [N,L,C]
  const float* Wqkv   = (const float*)d_in[1];   // [3C,C]
  const float* bqkv   = (const float*)d_in[2];   // [3C]
  const float* Aq     = (const float*)d_in[3];
  const float* Bq     = (const float*)d_in[4];
  const float* Ak     = (const float*)d_in[5];
  const float* Bk     = (const float*)d_in[6];
  const float* Av     = (const float*)d_in[7];
  const float* Bv     = (const float*)d_in[8];
  const float* lscale = (const float*)d_in[9];   // [H]
  const float* Wout   = (const float*)d_in[10];  // [C,C]
  const float* bout   = (const float*)d_in[11];  // [C]
  float*       out    = (float*)d_out;           // [N,L,C]

  char* ws = (char*)d_ws;
  size_t off = 0;
  auto alloc = [&](size_t bytes) {
    char* p = ws + off; off += (bytes + 255) & ~size_t(255); return p;
  };

  const size_t nx = (size_t)N_B * L_S * C_D;                 // 4M elems
  __bf16* xb    = (__bf16*)alloc(nx * 2);
  __bf16* wqkvb = (__bf16*)alloc((size_t)3 * C_D * C_D * 2); // merged qkv W
  __bf16* woutb = (__bf16*)alloc((size_t)C_D * C_D * 2);
  float*  qkvf  = (float*)alloc((size_t)N_B * L_S * 3 * C_D * 4);
  __bf16* Qn    = (__bf16*)alloc(nx * 2);
  __bf16* Kn    = (__bf16*)alloc(nx * 2);
  __bf16* Vt    = (__bf16*)alloc(nx * 2);
  __bf16* Ao    = (__bf16*)alloc(nx * 2);
  (void)ws_size;

  // 1) converts + LoRA fold
  k_convert_bf16<<<dim3(2048), dim3(256), 0, stream>>>(x, xb, (int)nx);
  k_convert_bf16<<<dim3(2048), dim3(256), 0, stream>>>(Wout, woutb, C_D * C_D);
  k_merge_lora<<<dim3((3 * C_D * C_D + 255) / 256), dim3(256), 0, stream>>>(
      Wqkv, Aq, Bq, Ak, Bk, Av, Bv, wqkvb);

  // 2) QKV GEMM: [4096,1024] x [3072,1024]^T -> [4096,3072]
  {
    dim3 grid(3 * C_D / BN, (N_B * L_S) / BM);
    k_gemm_bf16<<<grid, dim3(128), 0, stream>>>(xb, wqkvb, bqkv, qkvf,
                                                N_B * L_S, 3 * C_D, C_D);
  }

  // 3) split + l2norm + V transpose
  k_norm_split<<<dim3((N_B * L_S * H_N + 255) / 256), dim3(256), 0, stream>>>(
      qkvf, Qn, Kn, Vt);

  // 4) attention: one wave per (n,h, 16 q-rows)
  k_attention<<<dim3(N_B * H_N, L_S / 16), dim3(32), 0, stream>>>(
      Qn, Kn, Vt, lscale, Ao);

  // 5) out-proj GEMM: [4096,1024] x [1024,1024]^T + bias -> d_out
  {
    dim3 grid(C_D / BN, (N_B * L_S) / BM);
    k_gemm_bf16<<<grid, dim3(128), 0, stream>>>(Ao, woutb, bout, out,
                                                N_B * L_S, C_D, C_D);
  }
}
